// PointNetPP_77309412229
// MI455X (gfx1250) — matrix operand order repaired and checked by reading, not compiled, not run on hardware
//
#include <hip/hip_runtime.h>

#define DEVFN __device__ __forceinline__

typedef __attribute__((ext_vector_type(16))) __bf16 v16bf;
typedef __attribute__((ext_vector_type(8)))  float  v8f;

union FragU { v16bf v; uint4 q[2]; };

DEVFN unsigned short f2bf(float f) {
  unsigned u = __float_as_uint(f);
  u += 0x7FFFu + ((u >> 16) & 1u);   // round-to-nearest-even
  return (unsigned short)(u >> 16);
}
DEVFN float bf2f(unsigned short h) { return __uint_as_float(((unsigned)h) << 16); }

// A-fragment (16x32 bf16, MxK): lane half0 holds K {0..7,16..23}, half1 {8..15,24..31}
DEVFN v16bf load_fragA(const unsigned short* tile /*[16][64]*/, int lane, int kbase) {
  int half = (lane >> 4) & 1, r = lane & 15;
  const unsigned short* p = tile + r * 64 + kbase + half * 8;
  FragU t;
  t.q[0] = *(const uint4*)(p);        // elements 0..7
  t.q[1] = *(const uint4*)(p + 16);   // elements 8..15
  return t.v;
}
// B-fragment (32x16 bf16, KxN): lane n holds column n; half0 K {0..7,16..23}, half1 {8..15,24..31}
DEVFN v16bf load_fragB(const unsigned short* tile /*[16][64] rows=out-ch*/, int lane, int kbase) {
  int half = (lane >> 4) & 1, n = lane & 15;
  const unsigned short* p = tile + n * 64 + kbase + half * 8;
  FragU t;
  t.q[0] = *(const uint4*)(p);
  t.q[1] = *(const uint4*)(p + 16);
  return t.v;
}

// ---------------------------------------------------------------------------
// Templated WMMA MLP kernel: Out = act(A @ W^T + b), BN-affine fold on input,
// optional per-row mask + batch-stat accumulation. M must be a multiple of 128.
// AASYNC: stage the contiguous bf16 A tile via GLOBAL_LOAD_ASYNC_TO_LDS_B128.
// ---------------------------------------------------------------------------
template <int K, int COUT, int LDA, int LDC, bool ABF, bool AASYNC, bool OBF,
          bool FOLD, bool MASK, bool STATS, bool RELU>
__global__ __launch_bounds__(256) void k_mlp(
    const void* __restrict__ A, const float* __restrict__ W, const float* __restrict__ bias,
    const float* __restrict__ fold_s, const float* __restrict__ fold_t,
    const int* __restrict__ nbmask, void* __restrict__ Out, int M,
    float* __restrict__ ssum, float* __restrict__ ssq) {
  static_assert(!AASYNC || (ABF && LDA == 64), "async path needs contiguous bf16 rows");
  constexpr int NT = (COUT + 15) / 16;
  __shared__ __align__(16) unsigned short Ab[128 * 64];
  __shared__ __align__(16) unsigned short Wb[48 * 64];
  __shared__ float bb[48];
  __shared__ float ls[48], lq[48];
  __shared__ __align__(8) unsigned char msk[128];
  int tid = threadIdx.x;
  long row0 = (long)blockIdx.x * 128;

  // stage (folded) weights, bf16, zero-padded to 48x64
  for (int i = tid; i < 48 * 64; i += 256) {
    int o = i >> 6, k = i & 63;
    float w = 0.f;
    if (o < COUT && k < K) {
      w = W[o * K + k];
      if constexpr (FOLD) w *= fold_s[k];
    }
    Wb[i] = f2bf(w);
  }
  if (tid < 48) {
    float b = (tid < COUT) ? bias[tid] : 0.f;
    if constexpr (FOLD) {
      if (tid < COUT)
        for (int k = 0; k < K; ++k) b += W[tid * K + k] * fold_t[k];
    }
    bb[tid] = b;
    if constexpr (STATS) { ls[tid] = 0.f; lq[tid] = 0.f; }
  }
  if constexpr (MASK) {
    if (tid < 128) msk[tid] = (nbmask[row0 + tid] >= 0) ? 1 : 0;
  }

  // stage A tile
  if constexpr (AASYNC) {
    // contiguous 128x64 bf16 tile (16KB): per-lane async b128 copies to LDS
    unsigned long long gbase = (unsigned long long)((const char*)A + (size_t)row0 * (LDA * 2));
    unsigned ldsbase = (unsigned)(unsigned long long)(void*)Ab;
    unsigned ofs = (unsigned)tid * 64u;
#pragma unroll
    for (int j = 0; j < 4; ++j) {
      unsigned ldst = ldsbase + ofs + j * 16u;
      unsigned long long ga = gbase + ofs + j * 16u;
      asm volatile("global_load_async_to_lds_b128 %0, %1, off" ::"v"(ldst), "v"(ga) : "memory");
    }
    asm volatile("s_wait_asynccnt 0x0" ::: "memory");
  } else {
    for (int i = tid; i < 128 * 64; i += 256) {
      int r = i >> 6, k = i & 63;
      float v = 0.f;
      if (k < K) {
        long gr = row0 + r;
        if constexpr (ABF) v = bf2f(((const unsigned short*)A)[gr * LDA + k]);
        else               v = ((const float*)A)[gr * LDA + k];
      }
      Ab[i] = f2bf(v);
    }
  }
  __syncthreads();

  int wave = tid >> 5, lane = tid & 31;
  const unsigned short* Atile = Ab + wave * 16 * 64;
  v16bf fa0 = load_fragA(Atile, lane, 0);
  v16bf fa1 = load_fragA(Atile, lane, 32);
  int half = (lane >> 4) & 1;
  // each lane's 8 output rows are 8 consecutive mask bytes: one b64 LDS load
  unsigned long long mbits = ~0ull;
  if constexpr (MASK) mbits = *(const unsigned long long*)(msk + wave * 16 + half * 8);
#pragma unroll
  for (int nt = 0; nt < NT; ++nt) {
    v16bf fb0 = load_fragB(Wb + nt * 16 * 64, lane, 0);
    v8f acc = {0.f, 0.f, 0.f, 0.f, 0.f, 0.f, 0.f, 0.f};
    acc = __builtin_amdgcn_wmma_f32_16x16x32_bf16(false, fa0, false, fb0, (short)0, acc, false, false);
    if constexpr (K > 32) {
      v16bf fb1 = load_fragB(Wb + nt * 16 * 64, lane, 32);
      acc = __builtin_amdgcn_wmma_f32_16x16x32_bf16(false, fa1, false, fb1, (short)0, acc, false, false);
    }
    int col = nt * 16 + (lane & 15);
#pragma unroll
    for (int vr = 0; vr < 8; ++vr) {
      int r = wave * 16 + vr + half * 8;
      long gr = row0 + r;
      float v = acc[vr] + bb[col];
      if constexpr (RELU) v = fmaxf(v, 0.f);
      if (col < LDC) {
        float w = (col < COUT) ? v : 0.f;
        if constexpr (OBF) ((unsigned short*)Out)[gr * LDC + col] = f2bf(w);
        else               ((float*)Out)[gr * LDC + col] = w;
      }
      if constexpr (STATS) {
        if (col < COUT) {
          float mv = v;  // adding 0 for invalid rows is a no-op for the sums
          if constexpr (MASK) mv = ((mbits >> (8 * vr)) & 0xFFull) ? v : 0.f;
          atomicAdd(&ls[col], mv);
          atomicAdd(&lq[col], mv * mv);
        }
      }
    }
  }
  if constexpr (STATS) {
    __syncthreads();
    if (tid < COUT) {
      atomicAdd(&ssum[tid], ls[tid]);
      atomicAdd(&ssq[tid],  lq[tid]);
    }
  }
}

// ---------------------------------------------------------------------------
// SA layer-1 GEMM with fused edge-feature gather: A[row] = [x[nb], pos[nb]-q]
// H1 stored [m*64, 64] bf16 (cols 33..63 zeroed) so layer 2 can async-stage it.
// ---------------------------------------------------------------------------
__global__ __launch_bounds__(256) void k_sa_gemm1(
    const float* __restrict__ xl, const float* __restrict__ posl, const float* __restrict__ qpos,
    const int* __restrict__ nb, int m,
    const float* __restrict__ W /*[33,35]*/, const float* __restrict__ bias /*[33]*/,
    unsigned short* __restrict__ H1 /*[m*64,64]*/, float* __restrict__ ssum, float* __restrict__ ssq) {
  __shared__ __align__(16) unsigned short Ab[128 * 64];
  __shared__ __align__(16) unsigned short Wb[48 * 64];
  __shared__ float bb[48];
  __shared__ float ls[48], lq[48];
  __shared__ __align__(8) unsigned char msk[128];
  const int K = 35, Cout = 33;
  int tid = threadIdx.x;
  long row0 = (long)blockIdx.x * 128;

  for (int i = tid; i < 48 * 64; i += 256) {
    int o = i >> 6, k = i & 63;
    Wb[i] = f2bf((o < Cout && k < K) ? W[o * K + k] : 0.f);
  }
  if (tid < 48) { bb[tid] = (tid < Cout) ? bias[tid] : 0.f; ls[tid] = 0.f; lq[tid] = 0.f; }
  if (tid < 128) msk[tid] = (nb[row0 + tid] >= 0) ? 1 : 0;

  for (int i = tid; i < 128 * 64; i += 256) {
    int r = i >> 6, k = i & 63;
    long e = row0 + r;
    float v = 0.f;
    if (k < K) {
      int q = (int)(e >> 6);
      int p = nb[e];
      if (p >= 0) {
        if (k < 32) v = xl[p * 32 + k];
        else        v = posl[p * 3 + (k - 32)] - qpos[q * 3 + (k - 32)];
      }
    }
    Ab[i] = f2bf(v);
  }
  __syncthreads();

  int wave = tid >> 5, lane = tid & 31;
  const unsigned short* Atile = Ab + wave * 16 * 64;
  v16bf fa0 = load_fragA(Atile, lane, 0);
  v16bf fa1 = load_fragA(Atile, lane, 32);
  int half = (lane >> 4) & 1;
  unsigned long long mbits = *(const unsigned long long*)(msk + wave * 16 + half * 8);
#pragma unroll
  for (int nt = 0; nt < 3; ++nt) {
    v16bf fb0 = load_fragB(Wb + nt * 16 * 64, lane, 0);
    v16bf fb1 = load_fragB(Wb + nt * 16 * 64, lane, 32);
    v8f acc = {0.f, 0.f, 0.f, 0.f, 0.f, 0.f, 0.f, 0.f};
    acc = __builtin_amdgcn_wmma_f32_16x16x32_bf16(false, fa0, false, fb0, (short)0, acc, false, false);
    acc = __builtin_amdgcn_wmma_f32_16x16x32_bf16(false, fa1, false, fb1, (short)0, acc, false, false);
    int col = nt * 16 + (lane & 15);
#pragma unroll
    for (int vr = 0; vr < 8; ++vr) {
      int r = wave * 16 + vr + half * 8;
      long gr = row0 + r;
      float v = fmaxf(acc[vr] + bb[col], 0.f);
      float w = (col < Cout) ? v : 0.f;
      H1[gr * 64 + col] = f2bf(w);
      if (col < Cout) {
        float mv = ((mbits >> (8 * vr)) & 0xFFull) ? v : 0.f;
        atomicAdd(&ls[col], mv);
        atomicAdd(&lq[col], mv * mv);
      }
    }
  }
  // zero pad columns 48..63 so async restage + zero weights stay clean
  for (int i = tid; i < 128 * 16; i += 256) {
    int r = i >> 4, c = 48 + (i & 15);
    H1[(row0 + r) * 64 + c] = 0;
  }
  __syncthreads();
  if (tid < Cout) { atomicAdd(&ssum[tid], ls[tid]); atomicAdd(&ssq[tid], lq[tid]); }
}

// ---------------------------------------------------------------------------
// Farthest point sampling: single persistent workgroup, dmin resident in LDS.
// ---------------------------------------------------------------------------
__global__ void k_fps(const float* pos, int n, int m, int* out_idx) {
  extern __shared__ float sm[];
  float* dmin = sm;
  float* rv = sm + n;
  int* ri = (int*)(rv + blockDim.x);
  int tid = threadIdx.x, bs = blockDim.x;
  for (int i = tid; i < n; i += bs) dmin[i] = 1e30f;
  if (tid == 0) out_idx[0] = 0;
  __syncthreads();
  int last = 0;
  for (int it = 1; it < m; ++it) {
    float lx = pos[last * 3], ly = pos[last * 3 + 1], lz = pos[last * 3 + 2];
    float best = -1.f; int bi = 0;
    for (int i = tid; i < n; i += bs) {
      float dx = pos[i * 3] - lx, dy = pos[i * 3 + 1] - ly, dz = pos[i * 3 + 2] - lz;
      float d = dx * dx + dy * dy + dz * dz;
      float dm = fminf(dmin[i], d);
      dmin[i] = dm;
      if (dm > best) { best = dm; bi = i; }
    }
    rv[tid] = best; ri[tid] = bi;
    __syncthreads();
    for (int s = bs >> 1; s > 0; s >>= 1) {
      if (tid < s) {
        if (rv[tid + s] > rv[tid] || (rv[tid + s] == rv[tid] && ri[tid + s] < ri[tid])) {
          rv[tid] = rv[tid + s]; ri[tid] = ri[tid + s];
        }
      }
      __syncthreads();
    }
    last = ri[0];
    if (tid == 0) out_idx[it] = last;
    __syncthreads();
  }
}

__global__ void k_gather3(const float* src, const int* idx, float* dst, int m) {
  int i = blockIdx.x * blockDim.x + threadIdx.x;
  if (i < m * 3) { int r = i / 3, c = i % 3; dst[i] = src[idx[r] * 3 + c]; }
}

// ---------------------------------------------------------------------------
// Ball query: collect in-radius candidates in LDS, bitonic-select 64 nearest.
// ---------------------------------------------------------------------------
#define CAND 512
__global__ __launch_bounds__(256) void k_ballquery(const float* qpos, const float* pts,
                                                   int n, int m, int* nb, float* cntp) {
  __shared__ float cd[CAND];
  __shared__ int ci[CAND];
  __shared__ int cnt;
  int b = blockIdx.x, tid = threadIdx.x;
  if (tid == 0) cnt = 0;
  __syncthreads();
  float qx = qpos[b * 3], qy = qpos[b * 3 + 1], qz = qpos[b * 3 + 2];
  for (int i = tid; i < n; i += 256) {
    float dx = pts[i * 3] - qx, dy = pts[i * 3 + 1] - qy, dz = pts[i * 3 + 2] - qz;
    float d2 = dx * dx + dy * dy + dz * dz;
    if (d2 <= 4.0f) {
      int p = atomicAdd(&cnt, 1);
      if (p < CAND) { cd[p] = d2; ci[p] = i; }
    }
  }
  __syncthreads();
  int c = cnt; if (c > CAND) c = CAND;
  for (int i = c + tid; i < CAND; i += 256) { cd[i] = 1e30f; ci[i] = -1; }
  __syncthreads();
  for (int k = 2; k <= CAND; k <<= 1) {
    for (int j = k >> 1; j > 0; j >>= 1) {
      for (int i = tid; i < CAND; i += 256) {
        int l = i ^ j;
        if (l > i) {
          bool up = ((i & k) == 0);
          bool sw = up ? (cd[i] > cd[l]) : (cd[i] < cd[l]);
          if (sw) {
            float tf = cd[i]; cd[i] = cd[l]; cd[l] = tf;
            int ti = ci[i]; ci[i] = ci[l]; ci[l] = ti;
          }
        }
      }
      __syncthreads();
    }
  }
  if (tid < 64) nb[b * 64 + tid] = (tid < c) ? ci[tid] : -1;
  if (tid == 0) atomicAdd(cntp, (float)((c < 64) ? c : 64));
}

__global__ void k_zero(float* p, int n) {
  for (int i = blockIdx.x * blockDim.x + threadIdx.x; i < n; i += blockDim.x * gridDim.x) p[i] = 0.f;
}

// BN fold: s = g*rstd, t = be - s*mean   (pad channels zeroed)
__global__ void k_bn_fold(const float* ssum, const float* ssq, const float* cntp, float cnt_const,
                          const float* g, const float* be, int C, int Cpad, float* s, float* t) {
  int c = blockIdx.x * blockDim.x + threadIdx.x;
  if (c >= Cpad) return;
  if (c >= C) { s[c] = 0.f; t[c] = 0.f; return; }
  float cnt = cntp ? *cntp : cnt_const;
  float mean = ssum[c] / cnt;
  float var = ssq[c] / cnt - mean * mean;
  float rstd = rsqrtf(var + 1e-5f);
  float sc = g[c] * rstd;
  s[c] = sc; t[c] = be[c] - sc * mean;
}

// masked max-aggregation over K=64 neighbors with BN affine applied
__global__ __launch_bounds__(256) void k_sa_aggr(const unsigned short* H2, const int* nb,
                                                 const float* s, const float* t, int m, float* out) {
  int q = blockIdx.x * 8 + (threadIdx.x >> 5);
  int c = threadIdx.x & 31;
  if (q >= m) return;
  float best = -1e30f;
  for (int j = 0; j < 64; ++j) {
    if (nb[q * 64 + j] >= 0) {
      float v = s[c] * bf2f(H2[((long)q * 64 + j) * 32 + c]) + t[c];
      if (v > best) best = v;
    }
  }
  out[q * 32 + c] = best;
}

// ---------------------------------------------------------------------------
// knn-3 interpolation (one wave32 per point, butterfly top-3 merge) + concat skip
// ---------------------------------------------------------------------------
DEVFN void ins3(float d, int idx, float& d0, int& i0, float& d1, int& i1, float& d2, int& i2) {
  if (d < d0)      { d2 = d1; i2 = i1; d1 = d0; i1 = i0; d0 = d; i0 = idx; }
  else if (d < d1) { d2 = d1; i2 = i1; d1 = d;  i1 = idx; }
  else if (d < d2) { d2 = d;  i2 = idx; }
}

__global__ __launch_bounds__(256) void k_knn_interp(
    const float* pos_skip, int ns, const float* pos_cur, int nc,
    const float* x_cur /*[nc,32]*/, const float* x_skip /*[ns,32]*/,
    unsigned short* A /*[ns,64]*/) {
  int i = blockIdx.x * 8 + (threadIdx.x >> 5);
  int lane = threadIdx.x & 31;
  if (i >= ns) return;
  float px = pos_skip[i * 3], py = pos_skip[i * 3 + 1], pz = pos_skip[i * 3 + 2];
  float d0 = 1e30f, d1 = 1e30f, d2 = 1e30f;
  int i0 = 0, i1 = 0, i2 = 0;
  for (int p = lane; p < nc; p += 32) {
    float dx = pos_cur[p * 3] - px, dy = pos_cur[p * 3 + 1] - py, dz = pos_cur[p * 3 + 2] - pz;
    float d = dx * dx + dy * dy + dz * dz;
    ins3(d, p, d0, i0, d1, i1, d2, i2);
  }
  for (int off = 16; off > 0; off >>= 1) {
    float e0 = __shfl_xor(d0, off, 32), e1 = __shfl_xor(d1, off, 32), e2 = __shfl_xor(d2, off, 32);
    int j0 = __shfl_xor(i0, off, 32), j1 = __shfl_xor(i1, off, 32), j2 = __shfl_xor(i2, off, 32);
    ins3(e0, j0, d0, i0, d1, i1, d2, i2);
    ins3(e1, j1, d0, i0, d1, i1, d2, i2);
    ins3(e2, j2, d0, i0, d1, i1, d2, i2);
  }
  float w0 = 1.f / fmaxf(d0, 1e-16f), w1 = 1.f / fmaxf(d1, 1e-16f), w2 = 1.f / fmaxf(d2, 1e-16f);
  float ws = w0 + w1 + w2;
  int c = lane;
  float xi = (w0 * x_cur[i0 * 32 + c] + w1 * x_cur[i1 * 32 + c] + w2 * x_cur[i2 * 32 + c]) / ws;
  A[(long)i * 64 + c]      = f2bf(xi);
  A[(long)i * 64 + 32 + c] = f2bf(x_skip[i * 32 + c]);
}

// final BN affine (FP module output)
__global__ void k_affine(const unsigned short* H, const float* s, const float* t,
                         int n, int C, float* out) {
  int i = blockIdx.x * blockDim.x + threadIdx.x;
  if (i < n * C) { int c = i % C; out[i] = s[c] * bf2f(H[i]) + t[c]; }
}

// ---------------------------------------------------------------------------
extern "C" void kernel_launch(void* const* d_in, const int* in_sizes, int n_in,
                              void* d_out, int out_size, void* d_ws, size_t ws_size,
                              hipStream_t stream) {
  (void)in_sizes; (void)n_in; (void)out_size; (void)ws_size;
  const float* x    = (const float*)d_in[0];
  const float* pos0 = (const float*)d_in[1];
  // d_in[2] = norm: unused by PointConv path
  int i = 3;  // params flattened in jax pytree order (sorted dict keys)
  const float *fpW[3][2], *fpb[3][2], *fpbe[3][2], *fpg[3][2];
  for (int l = 0; l < 3; ++l)
    for (int j = 0; j < 2; ++j) {
      fpW[l][j]  = (const float*)d_in[i++];
      fpb[l][j]  = (const float*)d_in[i++];
      fpbe[l][j] = (const float*)d_in[i++];
      fpg[l][j]  = (const float*)d_in[i++];
    }
  const float* lin1W = (const float*)d_in[i++]; const float* lin1b = (const float*)d_in[i++];
  const float* lin2W = (const float*)d_in[i++]; const float* lin2b = (const float*)d_in[i++];
  const float* lin3W = (const float*)d_in[i++]; const float* lin3b = (const float*)d_in[i++];
  const float* lin4W = (const float*)d_in[i++]; const float* lin4b = (const float*)d_in[i++];
  const float *saW[3][2], *sab[3][2], *sabe[3][2], *sag[3][2];
  for (int l = 0; l < 3; ++l)
    for (int j = 0; j < 2; ++j) {
      saW[l][j]  = (const float*)d_in[i++];
      sab[l][j]  = (const float*)d_in[i++];
      sabe[l][j] = (const float*)d_in[i++];
      sag[l][j]  = (const float*)d_in[i++];
    }

  char* base = (char*)d_ws; size_t off = 0;
  auto carve = [&](size_t bytes) -> void* {
    void* p = base + off; off = (off + bytes + 255) & ~(size_t)255; return p;
  };
  float* h0   = (float*)carve((size_t)16384 * 32 * 4);
  float* pos1 = (float*)carve((size_t)8192 * 3 * 4);
  float* pos2 = (float*)carve((size_t)4096 * 3 * 4);
  float* pos3 = (float*)carve((size_t)2048 * 3 * 4);
  float* f1   = (float*)carve((size_t)8192 * 32 * 4);
  float* f2   = (float*)carve((size_t)4096 * 32 * 4);
  float* f3   = (float*)carve((size_t)2048 * 32 * 4);
  int*   idxb = (int*)carve((size_t)8192 * 4);
  int*   nb   = (int*)carve((size_t)8192 * 64 * 4);
  unsigned short* H1  = (unsigned short*)carve((size_t)524288 * 64 * 2);
  unsigned short* H2  = (unsigned short*)carve((size_t)524288 * 32 * 2);
  unsigned short* Afp = (unsigned short*)carve((size_t)16384 * 64 * 2);
  unsigned short* HH  = (unsigned short*)carve((size_t)16384 * 64 * 2);
  unsigned short* HH2 = (unsigned short*)carve((size_t)16384 * 32 * 2);
  float* xcA = (float*)carve((size_t)16384 * 32 * 4);
  float* xcB = (float*)carve((size_t)16384 * 32 * 4);
  float* th1 = (float*)carve((size_t)16384 * 32 * 4);
  float* th2 = (float*)carve((size_t)16384 * 32 * 4);
  float* st  = (float*)carve((size_t)1024 * 4);
  float* cnt = st;
  float *ssum1 = st + 64,  *ssq1 = st + 128, *s1 = st + 192, *t1 = st + 256;
  float *ssum2 = st + 320, *ssq2 = st + 384, *s2 = st + 448, *t2 = st + 512;

  // ---- lin1: h0 = relu(x @ W1^T + b1) ----
  k_mlp<16, 32, 16, 32, false, false, false, false, false, false, true>
      <<<dim3(128), dim3(256), 0, stream>>>(x, lin1W, lin1b, nullptr, nullptr, nullptr,
                                            h0, 16384, nullptr, nullptr);

  // ---- SA modules ----
  const float* posl = pos0; const float* featl = h0;
  float* posN[3]  = {pos1, pos2, pos3};
  float* featN[3] = {f1, f2, f3};
  int n = 16384;
  for (int l = 0; l < 3; ++l) {
    int m = n / 2;
    size_t smem = (size_t)n * 4 + 1024 * 8;
    k_fps<<<dim3(1), dim3(1024), smem, stream>>>(posl, n, m, idxb);
    k_gather3<<<dim3((m * 3 + 255) / 256), dim3(256), 0, stream>>>(posl, idxb, posN[l], m);
    k_zero<<<dim3(1), dim3(256), 0, stream>>>(st, 576);
    k_ballquery<<<dim3(m), dim3(256), 0, stream>>>(posN[l], posl, n, m, nb, cnt);
    k_sa_gemm1<<<dim3(m * 64 / 128), dim3(256), 0, stream>>>(
        featl, posl, posN[l], nb, m, saW[l][0], sab[l][0], H1, ssum1, ssq1);
    k_bn_fold<<<dim3(1), dim3(64), 0, stream>>>(ssum1, ssq1, cnt, 0.f,
                                                sag[l][0], sabe[l][0], 33, 48, s1, t1);
    k_mlp<33, 32, 64, 32, true, true, true, true, true, true, true>
        <<<dim3(m * 64 / 128), dim3(256), 0, stream>>>(H1, saW[l][1], sab[l][1], s1, t1, nb,
                                                       H2, m * 64, ssum2, ssq2);
    k_bn_fold<<<dim3(1), dim3(64), 0, stream>>>(ssum2, ssq2, cnt, 0.f,
                                                sag[l][1], sabe[l][1], 32, 32, s2, t2);
    k_sa_aggr<<<dim3((m + 7) / 8), dim3(256), 0, stream>>>(H2, nb, s2, t2, m, featN[l]);
    posl = posN[l]; featl = featN[l]; n = m;
  }

  // ---- FP modules ----
  const float* xskip[3] = {h0, f1, f2};
  const float* pskip[3] = {pos0, pos1, pos2};
  int nsz[3] = {16384, 8192, 4096};
  const float* xcur = f3; const float* poscur = pos3; int ncur = 2048;
  float* xcOut[3] = {xcA, xcB, xcA};
  int oi = 0;
  for (int lvl = 2; lvl >= 0; --lvl) {
    int ns = nsz[lvl];
    k_knn_interp<<<dim3((ns + 7) / 8), dim3(256), 0, stream>>>(
        pskip[lvl], ns, poscur, ncur, xcur, xskip[lvl], Afp);
    k_zero<<<dim3(1), dim3(256), 0, stream>>>(st, 576);
    k_mlp<64, 48, 64, 64, true, true, true, false, false, true, true>
        <<<dim3(ns / 128), dim3(256), 0, stream>>>(Afp, fpW[lvl][0], fpb[lvl][0], nullptr, nullptr,
                                                   nullptr, HH, ns, ssum1, ssq1);
    k_bn_fold<<<dim3(1), dim3(64), 0, stream>>>(ssum1, ssq1, (const float*)nullptr, (float)ns,
                                                fpg[lvl][0], fpbe[lvl][0], 48, 64, s1, t1);
    k_mlp<48, 32, 64, 32, true, true, true, true, false, true, true>
        <<<dim3(ns / 128), dim3(256), 0, stream>>>(HH, fpW[lvl][1], fpb[lvl][1], s1, t1, nullptr,
                                                   HH2, ns, ssum2, ssq2);
    k_bn_fold<<<dim3(1), dim3(64), 0, stream>>>(ssum2, ssq2, (const float*)nullptr, (float)ns,
                                                fpg[lvl][1], fpbe[lvl][1], 32, 32, s2, t2);
    k_affine<<<dim3((ns * 32 + 255) / 256), dim3(256), 0, stream>>>(HH2, s2, t2, ns, 32, xcOut[oi]);
    xcur = xcOut[oi]; poscur = pskip[lvl]; ncur = ns; ++oi;
  }

  // ---- head: lin2 -> relu -> lin3 -> relu -> lin4 ----
  k_mlp<32, 32, 32, 32, false, false, false, false, false, false, true>
      <<<dim3(128), dim3(256), 0, stream>>>(xcur, lin2W, lin2b, nullptr, nullptr, nullptr,
                                            th1, 16384, nullptr, nullptr);
  k_mlp<32, 32, 32, 32, false, false, false, false, false, false, true>
      <<<dim3(128), dim3(256), 0, stream>>>(th1, lin3W, lin3b, nullptr, nullptr, nullptr,
                                            th2, 16384, nullptr, nullptr);
  k_mlp<32, 2, 32, 2, false, false, false, false, false, false, false>
      <<<dim3(128), dim3(256), 0, stream>>>(th2, lin4W, lin4b, nullptr, nullptr, nullptr,
                                            d_out, 16384, nullptr, nullptr);
}
